// UNetDecoderLayer_16406775071033
// MI455X (gfx1250) — compile-verified, matrix operand order checked by reading, not run on hardware
//
#include <hip/hip_runtime.h>
#include <hip/hip_bf16.h>
#include <math.h>

typedef __attribute__((ext_vector_type(16))) __bf16 v16bf;
typedef __attribute__((ext_vector_type(8)))  __bf16 v8bf;
typedef __attribute__((ext_vector_type(4)))  __bf16 v4bf;
typedef __attribute__((ext_vector_type(8)))  float  v8f;

// ---- WMMA helpers (CDNA5 wave32, 16x16x32 bf16 -> f32) --------------------
__device__ __forceinline__ v8f wmma_bf16(v16bf a, v16bf b, v8f c) {
  return __builtin_amdgcn_wmma_f32_16x16x32_bf16(false, a, false, b, (short)0, c,
                                                 false, false);
}
__device__ __forceinline__ __bf16 tobf(float x) { return (__bf16)x; }

// Per-lane 16-bit operand fetch: for lane group lg the 16 elements are the
// contiguous bf16 runs [lg*8, lg*8+8) and [16+lg*8, 16+lg*8+8) of a 32-wide
// K-chunk stored row-major -> two ds_load_b128.
__device__ __forceinline__ v16bf ld_op(const __bf16* rowk0, int lg) {
  v8bf lo = *(const v8bf*)(rowk0 + lg * 8);
  v8bf hi = *(const v8bf*)(rowk0 + 16 + lg * 8);
  return __builtin_shufflevector(lo, hi, 0, 1, 2, 3, 4, 5, 6, 7, 8, 9, 10, 11,
                                 12, 13, 14, 15);
}
__device__ __forceinline__ v4bf pk4(float4 v) {
  return (v4bf){tobf(v.x), tobf(v.y), tobf(v.z), tobf(v.w)};
}

// ===========================================================================
// CPB table precompute: tbl[iter][z*100+y*10+x][head], 8 iters x 1000 x 4
// ===========================================================================
__global__ __launch_bounds__(128) void cpb_table_kernel(
    const float* __restrict__ w1, const float* __restrict__ b1,
    const float* __restrict__ w2, float* __restrict__ tbl) {
  int it = blockIdx.x, t = threadIdx.x;
  __shared__ float w1s[128][3];
  __shared__ float b1s[128];
  __shared__ float w2s[4][128];
  const float* w1p = w1 + (size_t)it * 128 * 3;
  for (int i = t; i < 384; i += 128) w1s[i / 3][i % 3] = w1p[i];
  b1s[t] = b1[it * 128 + t];
  for (int i = t; i < 512; i += 128) w2s[i >> 7][i & 127] = w2[(size_t)it * 512 + i];
  __syncthreads();
  const float step = 8.0f / 9.0f;
  for (int p = t; p < 1000; p += 128) {
    int zi = p / 100, yi = (p / 10) % 10, xi = p % 10;
    float cx = -4.f + step * zi, cy = -4.f + step * yi, cz = -4.f + step * xi;
    float a0 = 0, a1 = 0, a2 = 0, a3 = 0;
    for (int n = 0; n < 128; ++n) {
      float h = fmaxf(w1s[n][0] * cx + w1s[n][1] * cy + w1s[n][2] * cz + b1s[n], 0.f);
      a0 += h * w2s[0][n]; a1 += h * w2s[1][n];
      a2 += h * w2s[2][n]; a3 += h * w2s[3][n];
    }
    float* o = tbl + (size_t)(it * 1000 + p) * 4;
    o[0] = a0; o[1] = a1; o[2] = a2; o[3] = a3;
  }
}

// ===========================================================================
// LayerNorm over C=256; one wave per row.
// ===========================================================================
__global__ __launch_bounds__(256) void layernorm_kernel(
    const float* __restrict__ x, const float* __restrict__ g,
    const float* __restrict__ be, const float* __restrict__ qp,
    float* __restrict__ oln, float* __restrict__ oqk) {
  int t = threadIdx.x, lane = t & 31, w = t >> 5;
  int row = blockIdx.x * 8 + w;
  const float* xr = x + (size_t)row * 256;
  float v[8], s = 0.f, s2 = 0.f;
#pragma unroll
  for (int j = 0; j < 8; ++j) {
    v[j] = xr[lane + j * 32];
    s += v[j]; s2 += v[j] * v[j];
  }
#pragma unroll
  for (int m = 16; m >= 1; m >>= 1) {
    s += __shfl_xor(s, m, 32);
    s2 += __shfl_xor(s2, m, 32);
  }
  float mean = s * (1.f / 256.f);
  float var = s2 * (1.f / 256.f) - mean * mean;
  float rstd = rsqrtf(var + 1e-5f);
#pragma unroll
  for (int j = 0; j < 8; ++j) {
    int c = lane + j * 32;
    float y = (v[j] - mean) * rstd * g[c] + be[c];
    if (oln) oln[(size_t)row * 256 + c] = y;
    if (oqk) oqk[(size_t)row * 256 + c] = y + qp[(size_t)row * 256 + c];
  }
}

// ===========================================================================
// bf16-WMMA GEMM: C[M,N] = A[M,K] @ W[N,K]^T + bias (+res) (+relu)
// A tile stored [m][k], W tile stored [n][k] (operand-major; no transpose).
// Double-buffered LDS, one barrier per K-chunk.
// aMode==1: A rows gathered from memory laid out (4096, 8, 256).
// ===========================================================================
__global__ __launch_bounds__(256) void gemm_wmma(
    const float* __restrict__ A, const float* __restrict__ W,
    const float* __restrict__ bias, const float* __restrict__ res,
    float* __restrict__ Cout, int M, int N, int K, int aMode, int doRelu) {
  __shared__ __align__(16) __bf16 As[2][64][40];
  __shared__ __align__(16) __bf16 Ws[2][64][40];
  int t = threadIdx.x, lane = t & 31, w = t >> 5;
  int lm = lane & 15, lg = lane >> 4;
  int mbase = blockIdx.x * 64, nbase = blockIdx.y * 64;
  int mt0 = w >> 2, mt1 = 2 + (w >> 2), nt = w & 3;
  v8f acc0 = {}, acc1 = {};
  int col4 = (t & 7) * 4;
  int rbase = t >> 3;

  auto load_chunk = [&](int k0, int buf) {
#pragma unroll
    for (int i = 0; i < 2; ++i) {
      int row = rbase + i * 32;
      int grow = mbase + row;
      const float* ap;
      if (aMode == 0) {
        ap = A + (size_t)grow * K;
      } else {
        int bb = grow >> 12, kk = grow & 4095;
        ap = A + ((size_t)kk * 8 + bb) * 256;
      }
      *(v4bf*)&As[buf][row][col4] = pk4(*(const float4*)(ap + k0 + col4));
      *(v4bf*)&Ws[buf][row][col4] =
          pk4(*(const float4*)(W + (size_t)(nbase + row) * K + k0 + col4));
    }
  };

  load_chunk(0, 0);
  __syncthreads();
  int nchunks = K >> 5;
  for (int c = 0; c < nchunks; ++c) {
    if (c + 1 < nchunks) load_chunk((c + 1) << 5, (c + 1) & 1);
    int cur = c & 1;
    v16bf a0 = ld_op(&As[cur][mt0 * 16 + lm][0], lg);
    v16bf a1 = ld_op(&As[cur][mt1 * 16 + lm][0], lg);
    v16bf bv = ld_op(&Ws[cur][nt * 16 + lm][0], lg);
    acc0 = wmma_bf16(a0, bv, acc0);
    acc1 = wmma_bf16(a1, bv, acc1);
    __syncthreads();
  }
  int colg = nbase + nt * 16 + lm;
  float bvl = bias ? bias[colg] : 0.f;
#pragma unroll
  for (int r = 0; r < 8; ++r) {
    int row0 = mbase + mt0 * 16 + r + (lg << 3);
    int row1 = mbase + mt1 * 16 + r + (lg << 3);
    float v0 = acc0[r] + bvl, v1 = acc1[r] + bvl;
    if (res) {
      v0 += res[(size_t)row0 * N + colg];
      v1 += res[(size_t)row1 * N + colg];
    }
    if (doRelu) { v0 = fmaxf(v0, 0.f); v1 = fmaxf(v1, 0.f); }
    Cout[(size_t)row0 * N + colg] = v0;
    Cout[(size_t)row1 * N + colg] = v1;
  }
}

// ===========================================================================
// Self-attention: one WG per (b, head, 32-query tile); full S=256 in LDS.
// Qs[m][d], Ks[s][d] (operand-major for scores), VT[d][s] (operand-major for
// PV), probs converted once to bf16 Pb[m][s].
// ===========================================================================
__global__ __launch_bounds__(256) void self_attn_kernel(
    const float* __restrict__ qs, const float* __restrict__ ks,
    const float* __restrict__ vs, float* __restrict__ os) {
  __shared__ __align__(16) __bf16 Qs[32][72];
  __shared__ __align__(16) __bf16 Ks[256][72];
  __shared__ __align__(16) __bf16 VT[64][264];
  __shared__ __align__(16) __bf16 Pb[32][264];
  __shared__ float Sc[32][256];
  int t = threadIdx.x, lane = t & 31, w = t >> 5;
  int lm = lane & 15, lg = lane >> 4;
  int qt = blockIdx.x, h = blockIdx.y, b = blockIdx.z;

  for (int i = t; i < 32 * 16; i += 256) {
    int r = i >> 4, c4 = (i & 15) * 4;
    int q = qt * 32 + r;
    float4 qv = *(const float4*)(qs + ((size_t)q * 8 + b) * 256 + h * 64 + c4);
    qv.x *= 0.125f; qv.y *= 0.125f; qv.z *= 0.125f; qv.w *= 0.125f;
    *(v4bf*)&Qs[r][c4] = pk4(qv);
  }
  for (int i = t; i < 256 * 16; i += 256) {
    int s = i >> 4, c4 = (i & 15) * 4;
    *(v4bf*)&Ks[s][c4] =
        pk4(*(const float4*)(ks + ((size_t)s * 8 + b) * 256 + h * 64 + c4));
    float4 vv = *(const float4*)(vs + ((size_t)s * 8 + b) * 256 + h * 64 + c4);
    VT[c4 + 0][s] = tobf(vv.x); VT[c4 + 1][s] = tobf(vv.y);
    VT[c4 + 2][s] = tobf(vv.z); VT[c4 + 3][s] = tobf(vv.w);
  }
  __syncthreads();
  // scores: 2x16 tiles of 16x16, 4 per wave
#pragma unroll
  for (int i = 0; i < 4; ++i) {
    int id = w * 4 + i, mt = id >> 4, nt = id & 15;
    v8f s = {};
#pragma unroll
    for (int k0 = 0; k0 < 64; k0 += 32) {
      s = wmma_bf16(ld_op(&Qs[mt * 16 + lm][k0], lg),
                    ld_op(&Ks[nt * 16 + lm][k0], lg), s);
    }
#pragma unroll
    for (int r = 0; r < 8; ++r) Sc[mt * 16 + r + (lg << 3)][nt * 16 + lm] = s[r];
  }
  __syncthreads();
  if (t < 32) {
    float mx = -1e30f;
    for (int s = 0; s < 256; ++s) mx = fmaxf(mx, Sc[t][s]);
    float sum = 0.f;
    for (int s = 0; s < 256; ++s) sum += __expf(Sc[t][s] - mx);
    float inv = 1.f / sum;
    for (int s = 0; s < 256; ++s) Pb[t][s] = tobf(__expf(Sc[t][s] - mx) * inv);
  }
  __syncthreads();
  // probs @ v : 2x4 tiles, 1 per wave
  int mt = w >> 2, nt = w & 3;
  v8f o = {};
  for (int k0 = 0; k0 < 256; k0 += 32) {
    o = wmma_bf16(ld_op(&Pb[mt * 16 + lm][k0], lg),
                  ld_op(&VT[nt * 16 + lm][k0], lg), o);
  }
#pragma unroll
  for (int r = 0; r < 8; ++r) {
    int q = qt * 32 + mt * 16 + r + (lg << 3);
    os[((size_t)q * 8 + b) * 256 + h * 64 + nt * 16 + lm] = o[r];
  }
}

// ===========================================================================
// Fused cross-attention with on-the-fly rpe (flash-style online softmax).
// One WG per (b, 16-query tile); all 4 heads; K streamed in 128-key chunks.
// Ks[key][d] (scores B-operand), VT[d][key] (PV B-operand), Pb bf16 probs.
// ===========================================================================
__global__ __launch_bounds__(256) void cross_attn_kernel(
    const float* __restrict__ qc, const float* __restrict__ kc,
    const float* __restrict__ vc, const float* __restrict__ tblg,
    const float* __restrict__ refp, const float* __restrict__ xyz,
    float* __restrict__ xc) {
  __shared__ __align__(16) float tblS[32000];   // 8 iters x 1000 x 4 heads
  __shared__ __align__(16) __bf16 Ks[128][72];
  __shared__ __align__(16) __bf16 VT[64][136];
  __shared__ __align__(16) __bf16 Pb[4][16][136];
  __shared__ float Sc[4][16][128];
  __shared__ float rps[384];                    // refp tile [q][iter][xyz]
  __shared__ float rowm[64], rowl[64], rowscale[64];

  int t = threadIdx.x, lane = t & 31, w = t >> 5;
  int lm = lane & 15, lg = lane >> 4;
  int qt = blockIdx.x, b = blockIdx.y;

  for (int i = t; i < 32000; i += 256) tblS[i] = tblg[i];
  {
    size_t base = ((size_t)b * 256 + qt * 16) * 24;
    for (int i = t; i < 384; i += 256) rps[i] = refp[base + i];
  }
  if (t < 64) { rowm[t] = -1e30f; rowl[t] = 0.f; rowscale[t] = 0.f; }

  // q operand (per-wave head), kept in VGPRs for the whole kernel
  int h = w >> 1;
  v16bf qa0, qa1;
  {
    const float* qp = qc + (((size_t)(qt * 16 + lm)) * 8 + b) * 256 + h * 64;
#pragma unroll
    for (int e = 0; e < 8; ++e) {
      qa0[e]     = tobf(qp[lg * 8 + e] * 0.125f);
      qa0[e + 8] = tobf(qp[16 + lg * 8 + e] * 0.125f);
      qa1[e]     = tobf(qp[32 + lg * 8 + e] * 0.125f);
      qa1[e + 8] = tobf(qp[48 + lg * 8 + e] * 0.125f);
    }
  }
  v8f o0 = {}, o1 = {};
  int dt0 = (w & 1) * 2, dt1 = dt0 + 1;
  __syncthreads();

  for (int kb = 0; kb < 4096; kb += 128) {
    // stage K [key][d] and V transposed [d][key] as bf16
    for (int i = t; i < 128 * 16; i += 256) {
      int krel = i >> 4, c4 = (i & 15) * 4;
      size_t rowoff = ((size_t)(b * 4096 + kb + krel)) * 64 + c4;
      *(v4bf*)&Ks[krel][c4] = pk4(*(const float4*)(kc + rowoff));
      float4 vv = *(const float4*)(vc + rowoff);
      VT[c4 + 0][krel] = tobf(vv.x); VT[c4 + 1][krel] = tobf(vv.y);
      VT[c4 + 2][krel] = tobf(vv.z); VT[c4 + 3][krel] = tobf(vv.w);
    }
    __syncthreads();
    // WMMA scores: 4 heads x 8 n-tiles; wave w -> head w>>1, 4 tiles
#pragma unroll
    for (int i = 0; i < 4; ++i) {
      int nt = (w & 1) * 4 + i;
      v8f s = {};
      s = wmma_bf16(qa0, ld_op(&Ks[nt * 16 + lm][0], lg), s);
      s = wmma_bf16(qa1, ld_op(&Ks[nt * 16 + lm][32], lg), s);
#pragma unroll
      for (int r = 0; r < 8; ++r) Sc[h][r + (lg << 3)][nt * 16 + lm] = s[r];
    }
    __syncthreads();
    // rpe: each thread owns 8 (q,key) pairs; trilinear gives 4 heads at once
    for (int i = 0; i < 8; ++i) {
      int p = t + i * 256;
      int qrel = p >> 7, kr = p & 127;
      const float* xp = xyz + ((size_t)(b * 4096 + kb + kr)) * 3;
      float px = xp[0], py = xp[1], pz = xp[2];
      float r0 = 0.f, r1 = 0.f, r2 = 0.f, r3 = 0.f;
      for (int it = 0; it < 8; ++it) {
        const float* rp = &rps[(qrel * 8 + it) * 3];
        float dx = rp[0] - px, dy = rp[1] - py, dz = rp[2] - pz;
        const float inv3 = 1.0f / 3.0f;
        float gx = copysignf(__log2f(fabsf(dx) * 512.f + 1.f) * inv3, dx) * 0.25f;
        float gy = copysignf(__log2f(fabsf(dy) * 512.f + 1.f) * inv3, dy) * 0.25f;
        float gz = copysignf(__log2f(fabsf(dz) * 512.f + 1.f) * inv3, dz) * 0.25f;
        float fx = (gx + 1.f) * 4.5f, fy = (gy + 1.f) * 4.5f, fz = (gz + 1.f) * 4.5f;
        float xf = floorf(fx), yf = floorf(fy), zf = floorf(fz);
        float tx = fx - xf, ty = fy - yf, tz = fz - zf;
        int ix = (int)xf, iy = (int)yf, iz = (int)zf;
#pragma unroll
        for (int c = 0; c < 8; ++c) {
          int dxc = c & 1, dyc = (c >> 1) & 1, dzc = c >> 2;
          int xi = ix + dxc, yi = iy + dyc, zi = iz + dzc;
          if (xi >= 0 && xi < 10 && yi >= 0 && yi < 10 && zi >= 0 && zi < 10) {
            float wgt = (dxc ? tx : 1.f - tx) * (dyc ? ty : 1.f - ty) *
                        (dzc ? tz : 1.f - tz);
            const float4 tv =
                *(const float4*)&tblS[((it * 1000) + zi * 100 + yi * 10 + xi) * 4];
            r0 += tv.x * wgt; r1 += tv.y * wgt; r2 += tv.z * wgt; r3 += tv.w * wgt;
          }
        }
      }
      Sc[0][qrel][kr] += r0; Sc[1][qrel][kr] += r1;
      Sc[2][qrel][kr] += r2; Sc[3][qrel][kr] += r3;
    }
    __syncthreads();
    // online softmax per (head,q) row; write bf16 probs
    if (t < 64) {
      int hh = t >> 4, qrel = t & 15;
      float mold = rowm[t], mloc = -1e30f;
      for (int k = 0; k < 128; ++k) mloc = fmaxf(mloc, Sc[hh][qrel][k]);
      float mnew = fmaxf(mold, mloc);
      float corr = __expf(mold - mnew);
      float ssum = 0.f;
      for (int k = 0; k < 128; ++k) {
        float e = __expf(Sc[hh][qrel][k] - mnew);
        Pb[hh][qrel][k] = tobf(e);
        ssum += e;
      }
      rowl[t] = rowl[t] * corr + ssum;
      rowm[t] = mnew;
      rowscale[t] = corr;
    }
    __syncthreads();
    // rescale accumulators and accumulate probs @ V (K=128 -> 4 WMMA steps)
#pragma unroll
    for (int r = 0; r < 8; ++r) {
      float f = rowscale[h * 16 + r + (lg << 3)];
      o0[r] *= f; o1[r] *= f;
    }
#pragma unroll
    for (int k0 = 0; k0 < 128; k0 += 32) {
      v16bf av = ld_op(&Pb[h][lm][k0], lg);
      o0 = wmma_bf16(av, ld_op(&VT[dt0 * 16 + lm][k0], lg), o0);
      o1 = wmma_bf16(av, ld_op(&VT[dt1 * 16 + lm][k0], lg), o1);
    }
    __syncthreads();
  }
  // finalize: divide by l and store (rows q*B+b, head slice)
#pragma unroll
  for (int r = 0; r < 8; ++r) {
    int qrel = r + (lg << 3);
    float inv = 1.f / rowl[h * 16 + qrel];
    size_t base = (((size_t)(qt * 16 + qrel)) * 8 + b) * 256 + h * 64;
    xc[base + dt0 * 16 + lm] = o0[r] * inv;
    xc[base + dt1 * 16 + lm] = o1[r] * inv;
  }
}

// ===========================================================================
// Host launcher
// ===========================================================================
extern "C" void kernel_launch(void* const* d_in, const int* in_sizes, int n_in,
                              void* d_out, int out_size, void* d_ws, size_t ws_size,
                              hipStream_t stream) {
  const float* tgt      = (const float*)d_in[0];
  const float* memory   = (const float*)d_in[1];
  const float* refp     = (const float*)d_in[2];
  const float* xyz      = (const float*)d_in[3];
  const float* qpos     = (const float*)d_in[4];
  const float* sa_w     = (const float*)d_in[5];
  const float* sa_b     = (const float*)d_in[6];
  const float* sa_ow    = (const float*)d_in[7];
  const float* sa_ob    = (const float*)d_in[8];
  const float* wq       = (const float*)d_in[9];
  const float* bq       = (const float*)d_in[10];
  const float* wk       = (const float*)d_in[11];
  const float* bk       = (const float*)d_in[12];
  const float* wv       = (const float*)d_in[13];
  const float* bv       = (const float*)d_in[14];
  const float* wp       = (const float*)d_in[15];
  const float* bp       = (const float*)d_in[16];
  const float* cpb_w1   = (const float*)d_in[17];
  const float* cpb_b1   = (const float*)d_in[18];
  const float* cpb_w2   = (const float*)d_in[19];
  const float* g1       = (const float*)d_in[20];
  const float* be1      = (const float*)d_in[21];
  const float* g2       = (const float*)d_in[22];
  const float* be2      = (const float*)d_in[23];
  const float* g3       = (const float*)d_in[24];
  const float* be3      = (const float*)d_in[25];
  const float* w_ff1    = (const float*)d_in[26];
  const float* b_ff1    = (const float*)d_in[27];
  const float* w_ff2    = (const float*)d_in[28];
  const float* b_ff2    = (const float*)d_in[29];
  float* out = (float*)d_out;

  float* ws = (float*)d_ws;
  const size_t SLOT = 524288;           // 2048*256
  float* tbl = ws;                      // 32000 (padded to 32768)
  float* s[12];
  for (int i = 0; i < 12; ++i) s[i] = ws + 32768 + (size_t)i * SLOT;
  // lifetimes:
  //  s0 tgt2 -> later qc; s1 qk1 -> later kc (spans s1..s4);
  //  s2 qs -> later tgt_b; s3 ks -> later t3; s4 vs -> later hff;
  //  s5 os -> later xc; s6 tgt_a; s7 qk2; s8..s11 vc.

  dim3 gM(2048 / 64, 256 / 64);         // 2048x256 GEMMs
  dim3 gKV(32768 / 64, 1);              // memory projections 32768x64

  cpb_table_kernel<<<8, 128, 0, stream>>>(cpb_w1, cpb_b1, cpb_w2, tbl);

  // --- self attention block ---
  layernorm_kernel<<<256, 256, 0, stream>>>(tgt, g1, be1, qpos, s[0], s[1]);
  gemm_wmma<<<gM, 256, 0, stream>>>(s[1], sa_w,             sa_b,       nullptr, s[2], 2048, 256, 256, 0, 0); // q
  gemm_wmma<<<gM, 256, 0, stream>>>(s[1], sa_w + 256 * 256, sa_b + 256, nullptr, s[3], 2048, 256, 256, 0, 0); // k
  gemm_wmma<<<gM, 256, 0, stream>>>(s[0], sa_w + 512 * 256, sa_b + 512, nullptr, s[4], 2048, 256, 256, 0, 0); // v
  {
    dim3 gsa(8, 4, 8);
    self_attn_kernel<<<gsa, 256, 0, stream>>>(s[2], s[3], s[4], s[5]);
  }
  gemm_wmma<<<gM, 256, 0, stream>>>(s[5], sa_ow, sa_ob, tgt, s[6], 2048, 256, 256, 0, 0); // tgt_a

  // --- cross attention block ---
  layernorm_kernel<<<256, 256, 0, stream>>>(s[6], g2, be2, qpos, nullptr, s[7]);   // qk2
  gemm_wmma<<<gM, 256, 0, stream>>>(s[7], wq, bq, nullptr, s[0], 2048, 256, 256, 0, 0);   // qc
  gemm_wmma<<<gKV, 256, 0, stream>>>(memory, wk, bk, nullptr, s[1], 32768, 64, 256, 1, 0); // kc
  gemm_wmma<<<gKV, 256, 0, stream>>>(memory, wv, bv, nullptr, s[8], 32768, 64, 256, 1, 0); // vc
  {
    dim3 gx(16, 8);
    cross_attn_kernel<<<gx, 256, 0, stream>>>(s[0], s[1], s[8], tbl, refp, xyz, s[5]);
  }
  gemm_wmma<<<gM, 256, 0, stream>>>(s[5], wp, bp, s[6], s[2], 2048, 256, 256, 0, 0); // tgt_b

  // --- FFN block ---
  layernorm_kernel<<<256, 256, 0, stream>>>(s[2], g3, be3, nullptr, s[3], nullptr);
  gemm_wmma<<<gM, 256, 0, stream>>>(s[3], w_ff1, b_ff1, nullptr, s[4], 2048, 256, 256, 0, 1); // relu
  gemm_wmma<<<gM, 256, 0, stream>>>(s[4], w_ff2, b_ff2, s[2], out, 2048, 256, 256, 0, 0);

  (void)in_sizes; (void)n_in; (void)out_size; (void)ws_size;
}
